// GCN_14422500180192
// MI455X (gfx1250) — matrix operand order repaired and checked by reading, not compiled, not run on hardware
//
#include <hip/hip_runtime.h>

// ---------------------------------------------------------------------------
// GCN forward on gfx1250 (MI455X):
//   TDM(tensor_load_to_lds)-staged, register-blocked bf16 WMMA GEMMs
//   + atomic-scatter SpMM + fused bias/leakyrelu/dropout tail.
// ---------------------------------------------------------------------------

#define NEG_SLOPE 0.01f
#define KEEP_P    0.7f               // 1 - dropout_p
#define KEEP_THRESH 3006477107u      // floor(0.7 * 2^32)

typedef __attribute__((ext_vector_type(16))) __bf16    v16bf;
typedef __attribute__((ext_vector_type(8)))  float     v8f;
typedef __attribute__((ext_vector_type(4)))  unsigned  v4u;
typedef __attribute__((ext_vector_type(8)))  int       v8i;
typedef __attribute__((ext_vector_type(4)))  int       v4i;

// ---- packed f32 -> 2x bf16 (RNE), 3 VALU ops per pair ----------------------
__device__ __forceinline__ unsigned rne_hi16(float f) {
  unsigned u = __builtin_bit_cast(unsigned, f);
  return u + (0x7FFFu + ((u >> 16) & 1u));
}
__device__ __forceinline__ unsigned pack_bf2(float lo, float hi) {
  return (rne_hi16(lo) >> 16) | (rne_hi16(hi) & 0xFFFF0000u);
}

__device__ __forceinline__ unsigned hash_u32(unsigned x) {
  x ^= x >> 16; x *= 0x7feb352du;
  x ^= x >> 15; x *= 0x846ca68bu;
  x ^= x >> 16;
  return x;
}
// deterministic hash dropout: keep w.p. 0.7, scale 1/0.7
__device__ __forceinline__ float drop_scale(unsigned row, unsigned col, unsigned seed) {
  unsigned h = hash_u32(seed ^ (row * 2654435761u + col * 40503u + 0x9e3779b9u));
  return (h <= KEEP_THRESH) ? (1.0f / KEEP_P) : 0.0f;
}

// ---------------------------------------------------------------------------
// TDM: async-load one 16-row x 32-col f32 A-tile (row stride K elems) to LDS.
// D# per CDNA5 ISA (2-D tensor, groups 2/3 zero).
// This toolchain exposes the 6-arg builtin: (g0, g1, g2, g3, g4, cpol).
// ---------------------------------------------------------------------------
__device__ __forceinline__ void tdm_load_a_tile(const float* gtile, int K,
                                                unsigned lds_byte_addr) {
  unsigned long long ga = (unsigned long long)gtile;
  v4u g0;
  g0.x = 1u;                                          // count=1, user mode
  g0.y = lds_byte_addr;                               // lds_addr
  g0.z = (unsigned)(ga & 0xFFFFFFFFull);              // global_addr[31:0]
  g0.w = (unsigned)((ga >> 32) & 0x1FFFFFFull)        // global_addr[56:32]
       | (2u << 30);                                  // type=2 ("image")
  v8i g1;
  g1[0] = 0x00020000;                 // data_size=2 -> 4-byte elements
  g1[1] = (K & 0xFFFF) << 16;         // tensor_dim0[15:0]
  g1[2] = (K >> 16) | (16 << 16);     // tensor_dim0[31:16] | tensor_dim1=16
  g1[3] = (32 << 16);                 // tile_dim0 = 32 (K-chunk)
  g1[4] = 16;                         // tile_dim1 = 16 rows, tile_dim2 = 0
  g1[5] = K;                          // tensor_dim0_stride[31:0] = K elems
  g1[6] = 0;                          // stride hi / dim1_stride lo
  g1[7] = 0;
  v4i z4 = {0, 0, 0, 0};
  v8i z8 = {0, 0, 0, 0, 0, 0, 0, 0};
  __builtin_amdgcn_tensor_load_to_lds(g0, g1, z4, z4, z8, 0); // TENSORcnt++
}

// ---------------------------------------------------------------------------
// Pre-pack B[K,N] (f32 row-major) into per-wave WMMA bf16 fragments:
// layout [ntile][kstep][lane] x 16 bf16 (32 B contiguous per lane).
// Lane n=lane&15, khalf=lane>>4 holds B[k0+khalf*16 + j][ntile*16+n], j=0..15.
// ---------------------------------------------------------------------------
__global__ void pack_b_frags(const float* __restrict__ B, unsigned* __restrict__ F,
                             int K, int N)
{
  const int ksteps = K >> 5;
  const int total  = (N >> 4) * ksteps * 32;
  const int tid = blockIdx.x * blockDim.x + threadIdx.x;
  if (tid >= total) return;
  const int lane = tid & 31;
  const int frag = tid >> 5;
  const int nt = frag / ksteps;
  const int ks = frag % ksteps;
  const int n  = nt * 16 + (lane & 15);
  const int k0 = ks * 32 + (lane >> 4) * 16;
  unsigned* dst = F + (size_t)tid * 8;
#pragma unroll
  for (int j = 0; j < 8; ++j) {
    float lo = B[(size_t)(k0 + 2 * j)     * N + n];
    float hi = B[(size_t)(k0 + 2 * j + 1) * N + n];
    dst[j] = pack_bf2(lo, hi);
  }
}

// ---------------------------------------------------------------------------
// C[M,N] = pre(A)[M,K] @ B, A staged by TDM (double-buffered LDS), B from
// pre-packed bf16 fragments. One wave computes a 16 x 64 strip: 4 back-to-
// back WMMAs per k-step (compile-time, no inter-WMMA branches).
// MODE 0: identity, 1: lrelu(a+bias[k]), 2: dropout(lrelu(a+bias[k])).
// ---------------------------------------------------------------------------
template <int MODE>
__global__ void __launch_bounds__(256)
gemm_tdm_wmma(const float* __restrict__ A,
              const __bf16* __restrict__ Bfrag,
              float* __restrict__ C,
              int M, int K, int N,
              const float* __restrict__ bias,
              unsigned seed)
{
  constexpr int NT = 4;
  __shared__ float atile[8][2][512];          // 8 waves x double-buffer x 16x32
  const int lane = threadIdx.x & 31;
  const int wv   = threadIdx.x >> 5;
  const int wave = blockIdx.x * (blockDim.x >> 5) + wv;
  const int mtiles = M >> 4, ntiles = N >> 4;
  const int ngroups = ntiles / NT;
  if (wave >= mtiles * ngroups) return;
  const int mt = wave / ngroups;
  const int ng = wave % ngroups;
  const int ksteps = K >> 5;

  const int m     = lane & 15;     // A row / B,D column within tile
  const int khalf = lane >> 4;
  const unsigned row = (unsigned)(mt * 16 + m);

  const unsigned ldsb = (unsigned)(unsigned long long)&atile[wv][0][0];
  const float* tileA = A + (long long)mt * 16 * K;

  tdm_load_a_tile(tileA, K, ldsb);                    // prologue: ks=0 -> buf0

  v8f acc0 = {}, acc1 = {}, acc2 = {}, acc3 = {};

  for (int ks = 0; ks < ksteps; ++ks) {
    // pipeline: kick off next tile's DMA, then wait for current buffer only
    if (ks + 1 < ksteps) {
      tdm_load_a_tile(tileA + (ks + 1) * 32, K,
                      ldsb + (unsigned)((ks + 1) & 1) * 2048u);
      __builtin_amdgcn_s_wait_tensorcnt(1);
    } else {
      __builtin_amdgcn_s_wait_tensorcnt(0);
    }

    // A fragment from LDS (2x 16B + 2x 16B contiguous ds reads per lane)
    const float* ar = &atile[wv][ks & 1][0] + m * 32;
    const int c1 = khalf * 16;
    const int c2 = khalf * 16 + 8;
    float av[16];
#pragma unroll
    for (int j = 0; j < 8; ++j) { av[j] = ar[c1 + j]; av[8 + j] = ar[c2 + j]; }
    if constexpr (MODE >= 1) {
#pragma unroll
      for (int j = 0; j < 8; ++j) {
        const int k1 = ks * 32 + c1 + j, k2 = ks * 32 + c2 + j;
        float f1 = av[j]     + bias[k1]; f1 = (f1 >= 0.f) ? f1 : NEG_SLOPE * f1;
        float f2 = av[8 + j] + bias[k2]; f2 = (f2 >= 0.f) ? f2 : NEG_SLOPE * f2;
        if constexpr (MODE == 2) {
          f1 *= drop_scale(row, (unsigned)k1, seed);
          f2 *= drop_scale(row, (unsigned)k2, seed);
        }
        av[j] = f1; av[8 + j] = f2;
      }
    }
    union { v16bf v; unsigned u[8]; } af;
#pragma unroll
    for (int j = 0; j < 8; ++j) af.u[j] = pack_bf2(av[2 * j], av[2 * j + 1]);

    // 4 x WMMA, B fragments are contiguous 32B-per-lane loads
#define BFR(t) (*(const v16bf*)(Bfrag + \
        ((size_t)((ng * NT + (t)) * ksteps + ks) * 32 + lane) * 16))
    acc0 = __builtin_amdgcn_wmma_f32_16x16x32_bf16(false, af.v, false, BFR(0),
                                                   (short)0, acc0, false, false);
    acc1 = __builtin_amdgcn_wmma_f32_16x16x32_bf16(false, af.v, false, BFR(1),
                                                   (short)0, acc1, false, false);
    acc2 = __builtin_amdgcn_wmma_f32_16x16x32_bf16(false, af.v, false, BFR(2),
                                                   (short)0, acc2, false, false);
    acc3 = __builtin_amdgcn_wmma_f32_16x16x32_bf16(false, af.v, false, BFR(3),
                                                   (short)0, acc3, false, false);
#undef BFR
  }

  // D layout: VGPR v -> row khalf*8 + v, col = lane&15
  float* cbase = C + (long long)(mt * 16 + khalf * 8) * N + m;
#pragma unroll
  for (int v = 0; v < 8; ++v) {
    const long long ro = (long long)v * N;
    cbase[ro + (ng * NT + 0) * 16] = acc0[v];
    cbase[ro + (ng * NT + 1) * 16] = acc1[v];
    cbase[ro + (ng * NT + 2) * 16] = acc2[v];
    cbase[ro + (ng * NT + 3) * 16] = acc3[v];
  }
}

// ---------------------------------------------------------------------------
// Y[rows[e], :] += vals[e] * X[cols[e], :]   (COO scatter-add, d % 4 == 0)
// ---------------------------------------------------------------------------
__global__ void spmm_scatter(const float* __restrict__ vals,
                             const int* __restrict__ rows,
                             const int* __restrict__ cols,
                             const float* __restrict__ X,
                             float* __restrict__ Y,
                             int E, int d)
{
  const int chunks = d >> 2;
  const long long total  = (long long)E * chunks;
  const long long stride = (long long)gridDim.x * blockDim.x;
  for (long long idx = (long long)blockIdx.x * blockDim.x + threadIdx.x;
       idx < total; idx += stride) {
    if (idx + stride < total) {                       // hide index latency
      __builtin_prefetch(cols + (idx + stride) / chunks, 0, 0);
      __builtin_prefetch(rows + (idx + stride) / chunks, 0, 0);
    }
    const int e = (int)(idx / chunks);
    const int c = (int)(idx % chunks);
    const float v = vals[e];
    const float4 xv = *(const float4*)(X + (long long)cols[e] * d + c * 4);
    float* yp = Y + (long long)rows[e] * d + c * 4;
    atomicAdd(yp + 0, v * xv.x);
    atomicAdd(yp + 1, v * xv.y);
    atomicAdd(yp + 2, v * xv.z);
    atomicAdd(yp + 3, v * xv.w);
  }
}

__global__ void zero_f32(float* __restrict__ p, long long n) {
  const long long stride = (long long)gridDim.x * blockDim.x;
  for (long long i = (long long)blockIdx.x * blockDim.x + threadIdx.x; i < n; i += stride)
    p[i] = 0.f;
}

__global__ void zero_out(float* o) { o[0] = 0.f; }

// ---------------------------------------------------------------------------
// out += mean_i( dropout(lrelu(H3[i,:]+lb1)) . lw2 + lb2 ),  H3: [n,64]
// ---------------------------------------------------------------------------
__global__ void mlp_tail_mean(const float* __restrict__ H3,
                              const float* __restrict__ lb1,
                              const float* __restrict__ lw2,
                              const float* __restrict__ lb2,
                              float* __restrict__ out,
                              int n, unsigned seed)
{
  float local = 0.f;
  const long long stride = (long long)gridDim.x * blockDim.x;
  for (long long i = (long long)blockIdx.x * blockDim.x + threadIdx.x; i < n; i += stride) {
    const float* h = H3 + i * 64;
    float acc = lb2[0];
#pragma unroll
    for (int j = 0; j < 64; ++j) {
      float v = h[j] + lb1[j];
      v = (v >= 0.f) ? v : NEG_SLOPE * v;
      v *= drop_scale((unsigned)i, (unsigned)j, seed);
      acc += v * lw2[j];
    }
    local += acc;
  }
  __shared__ float red[256];
  red[threadIdx.x] = local;
  __syncthreads();
  for (int s = blockDim.x >> 1; s > 0; s >>= 1) {
    if ((int)threadIdx.x < s) red[threadIdx.x] += red[threadIdx.x + s];
    __syncthreads();
  }
  if (threadIdx.x == 0) atomicAdd(out, red[0] / (float)n);
}

// ---------------------------------------------------------------------------
extern "C" void kernel_launch(void* const* d_in, const int* in_sizes, int n_in,
                              void* d_out, int out_size, void* d_ws, size_t ws_size,
                              hipStream_t stream)
{
  const float* x    = (const float*)d_in[0];
  const int*   rows = (const int*)  d_in[1];
  const int*   cols = (const int*)  d_in[2];
  const float* vals = (const float*)d_in[3];
  const float* W1   = (const float*)d_in[4];
  const float* b1   = (const float*)d_in[5];
  const float* W2   = (const float*)d_in[6];
  const float* b2   = (const float*)d_in[7];
  const float* lw1  = (const float*)d_in[8];
  const float* lb1  = (const float*)d_in[9];
  const float* lw2  = (const float*)d_in[10];
  const float* lb2  = (const float*)d_in[11];

  const int F = 256, C1 = 128, C2 = 64, H = 64;
  const int N = in_sizes[0] / F;      // 100000 (multiple of 16)
  const int E = in_sizes[3];          // 3200000

  // workspace: 2 ping-pong activation buffers + bf16 weight fragments
  float*  bufA   = (float*)d_ws;                       // N*C1 f32
  float*  bufB   = bufA + (size_t)N * C1;              // N*C1 f32
  __bf16* fragW1 = (__bf16*)(bufB + (size_t)N * C1);   // 8*8 *512 bf16
  __bf16* fragW2 = fragW1 + (size_t)(C1/16)*(F/32)*512;   // 4*4 *512
  __bf16* fragL1 = fragW2 + (size_t)(C2/16)*(C1/32)*512;  // 4*2 *512

  const int BLK = 256, WPB = BLK / 32;

  // ---- pre-pack weights into WMMA bf16 fragment layout (tiny, reused) ----
  pack_b_frags<<<8, BLK, 0, stream>>>(W1,  (unsigned*)fragW1, F,  C1);
  pack_b_frags<<<2, BLK, 0, stream>>>(W2,  (unsigned*)fragW2, C1, C2);
  pack_b_frags<<<1, BLK, 0, stream>>>(lw1, (unsigned*)fragL1, C2, H);

  // 1) XW1 = x @ W1                              -> bufA [N,128]
  {
    const int waves = (N / 16) * ((C1 / 16) / 4);
    gemm_tdm_wmma<0><<<(waves + WPB - 1) / WPB, BLK, 0, stream>>>(
        x, fragW1, bufA, N, F, C1, nullptr, 0u);
  }
  // 2) AGG1 = spmm(adj, XW1)                     -> bufB [N,128]
  zero_f32<<<2048, BLK, 0, stream>>>(bufB, (long long)N * C1);
  spmm_scatter<<<4096, BLK, 0, stream>>>(vals, rows, cols, bufA, bufB, E, C1);

  // 3) HW2 = lrelu(AGG1 + b1) @ W2               -> bufA [N,64]
  {
    const int waves = (N / 16) * ((C2 / 16) / 4);
    gemm_tdm_wmma<1><<<(waves + WPB - 1) / WPB, BLK, 0, stream>>>(
        bufB, fragW2, bufA, N, C1, C2, b1, 0u);
  }
  // 4) AGG2 = spmm(adj, HW2)                     -> bufB [N,64]
  zero_f32<<<2048, BLK, 0, stream>>>(bufB, (long long)N * C2);
  spmm_scatter<<<4096, BLK, 0, stream>>>(vals, rows, cols, bufA, bufB, E, C2);

  // 5) H3 = dropout(lrelu(AGG2 + b2)) @ lw1      -> bufA [N,64]
  {
    const int waves = (N / 16) * ((H / 16) / 4);
    gemm_tdm_wmma<2><<<(waves + WPB - 1) / WPB, BLK, 0, stream>>>(
        bufB, fragL1, bufA, N, C2, H, b2, 0x12345697u);
  }
  // 6) out = mean( dropout(lrelu(H3 + lb1)) @ lw2 + lb2 )
  zero_out<<<1, 1, 0, stream>>>((float*)d_out);
  mlp_tail_mean<<<1024, BLK, 0, stream>>>(bufA, lb1, lw2, lb2, (float*)d_out,
                                          N, 0x89abcdefu);
}
